// Mamba3Block_46651934769490
// MI455X (gfx1250) — compile-verified, matrix operand order checked by reading, not run on hardware
//
#include <hip/hip_runtime.h>

#define BATCHN  2
#define SEQL    4096
#define DMODEL  768
#define DINNER  1536
#define NHEADSN 24
#define HEADD   64
#define DSTATE  128
#define CHUNKL  32
#define NCHUNK  (SEQL / CHUNKL)      // 128
#define CONVDIM 1792
#define DPROJ   3352
#define DPROJP  3456                 // padded: multiple of 128 for GEMM block tile
#define BL      (BATCHN * SEQL)      // 8192
#define LDSTR   40                   // padded LDS row stride (halfs) for GEMM tiles
#define DSTP    (DSTATE + 4)         // padded fp32 LDS row (132 dwords) for SSD tiles

typedef __bf16 bf16x16 __attribute__((ext_vector_type(16)));
typedef float  f32x8   __attribute__((ext_vector_type(8)));
typedef unsigned int u32x4 __attribute__((ext_vector_type(4)));
typedef unsigned int u32x8 __attribute__((ext_vector_type(8)));

union Frag  { bf16x16 v; uint4 u[2]; };
union FragS { bf16x16 v; unsigned short h[16]; };

static __device__ __forceinline__ unsigned short f32_to_bf16(float f) {
    unsigned int u = __float_as_uint(f);
    u += 0x7fffu + ((u >> 16) & 1u);          // round to nearest even
    return (unsigned short)(u >> 16);
}

static __device__ __forceinline__ float siluf(float v) {
    return v / (1.0f + __expf(-v));
}

// ---------------------------------------------------------------- rmsnorm(pre) -> bf16
__global__ void k_rmsnorm_in(const float* __restrict__ x, const float* __restrict__ scale,
                             unsigned short* __restrict__ yb) {
    int row = blockIdx.x;
    const float* xr = x + (size_t)row * DMODEL;
    __shared__ float red[8];
    float ss = 0.f;
    for (int d = threadIdx.x; d < DMODEL; d += 256) { float v = xr[d]; ss += v * v; }
    for (int o = 16; o > 0; o >>= 1) ss += __shfl_down(ss, o, 32);
    int wid = threadIdx.x >> 5, lane = threadIdx.x & 31;
    if (lane == 0) red[wid] = ss;
    __syncthreads();
    if (threadIdx.x == 0) {
        float t = 0.f;
        for (int i = 0; i < 8; i++) t += red[i];
        red[0] = rsqrtf(t / (float)DMODEL + 1e-6f);
    }
    __syncthreads();
    float r = red[0];
    for (int d = threadIdx.x; d < DMODEL; d += 256)
        yb[(size_t)row * DMODEL + d] = f32_to_bf16(xr[d] * r * (1.0f + scale[d]));
}

// ---------------------------------------------------------------- weight prep (transpose + bf16)
__global__ void k_prep_win(const float* __restrict__ W, unsigned short* __restrict__ WT) {
    int idx = blockIdx.x * 256 + threadIdx.x;          // over DPROJP * DMODEL
    if (idx >= DPROJP * DMODEL) return;
    int n = idx / DMODEL, k = idx % DMODEL;
    float v = (n < DPROJ) ? W[(size_t)k * DPROJ + n] : 0.f;
    WT[idx] = f32_to_bf16(v);
}

__global__ void k_prep_wout(const float* __restrict__ W, unsigned short* __restrict__ WT) {
    int idx = blockIdx.x * 256 + threadIdx.x;          // over DMODEL * DINNER
    if (idx >= DMODEL * DINNER) return;
    int n = idx / DINNER, k = idx % DINNER;
    WT[idx] = f32_to_bf16(W[(size_t)k * DMODEL + n]);
}

// ---------------------------------------------------------------- bf16 WMMA GEMM, async-LDS staged
// C[M,N] = A[M,K] * Bt^T, Bt is [N,K] row-major. Block: 256 thr = 8 waves (2M x 4N),
// block tile 64M x 128N, wave tile 32x32 (2x2 WMMA frags). Double-buffered LDS filled
// with global_load_async_to_lds_b128 (ASYNCcnt). M%64==0, N%128==0, K%32==0.
__global__ void k_gemm_bf16(const unsigned short* __restrict__ A,
                            const unsigned short* __restrict__ Bt,
                            float* __restrict__ C,
                            int M, int N, int K) {
    __shared__ unsigned short Ab[2][64 * LDSTR];
    __shared__ unsigned short Bb[2][128 * LDSTR];

    int t = threadIdx.x;
    int w = t >> 5, lane = t & 31;
    int mi = w >> 2, ni = w & 3;
    int mblk = blockIdx.y * 64;
    int nblk = blockIdx.x * 128;

    // async-copy assignment: A = 64 rows x 4 chunks(16B), B = 128 rows x 2x(32B)
    int arow = t >> 2, achk = t & 3;
    const unsigned short* agp = A + (size_t)(mblk + arow) * K + achk * 8;
    unsigned alds = (unsigned)(size_t)&Ab[0][arow * LDSTR + achk * 8];
    int brow = t >> 1, bchk = (t & 1) * 2;
    const unsigned short* bgp = Bt + (size_t)(nblk + brow) * K + bchk * 8;
    unsigned blds = (unsigned)(size_t)&Bb[0][brow * LDSTR + bchk * 8];
    const unsigned bufA = 64 * LDSTR * 2;     // bytes between double buffers
    const unsigned bufB = 128 * LDSTR * 2;

    auto issue_tile = [&](int buf, int kt) {
        unsigned long long ag = (unsigned long long)(agp + (size_t)kt * 32);
        unsigned long long bg = (unsigned long long)(bgp + (size_t)kt * 32);
        unsigned al = alds + (unsigned)buf * bufA;
        unsigned bl = blds + (unsigned)buf * bufB;
        asm volatile("global_load_async_to_lds_b128 %0, %1, off"
                     :: "v"(al), "v"(ag) : "memory");
        asm volatile("global_load_async_to_lds_b128 %0, %1, off"
                     :: "v"(bl), "v"(bg) : "memory");
        asm volatile("global_load_async_to_lds_b128 %0, %1, off offset:16"
                     :: "v"(bl), "v"(bg) : "memory");
    };

    int r = lane & 15;
    int koff = (lane >> 4) << 3;              // halfs
    const unsigned short* abase = &Ab[0][(mi * 32 + r) * LDSTR + koff];
    const unsigned short* bbase = &Bb[0][(ni * 32 + r) * LDSTR + koff];

    f32x8 acc[2][2] = {};
    int nk = K >> 5;
    issue_tile(0, 0);
    for (int kt = 0; kt < nk; kt++) {
        int cur = kt & 1;
        if (kt + 1 < nk) {
            issue_tile(1 - cur, kt + 1);
            asm volatile("s_wait_asynccnt 0x3" ::: "memory");   // current tile complete
        } else {
            asm volatile("s_wait_asynccnt 0x0" ::: "memory");
        }
        __syncthreads();
        const unsigned short* ab = abase + cur * (64 * LDSTR);
        const unsigned short* bb = bbase + cur * (128 * LDSTR);
        Frag a0, a1, b0, b1;
        a0.u[0] = *(const uint4*)(ab);               a0.u[1] = *(const uint4*)(ab + 16);
        a1.u[0] = *(const uint4*)(ab + 16 * LDSTR);  a1.u[1] = *(const uint4*)(ab + 16 * LDSTR + 16);
        b0.u[0] = *(const uint4*)(bb);               b0.u[1] = *(const uint4*)(bb + 16);
        b1.u[0] = *(const uint4*)(bb + 16 * LDSTR);  b1.u[1] = *(const uint4*)(bb + 16 * LDSTR + 16);
        acc[0][0] = __builtin_amdgcn_wmma_f32_16x16x32_bf16(false, a0.v, false, b0.v, (short)0, acc[0][0], false, false);
        acc[0][1] = __builtin_amdgcn_wmma_f32_16x16x32_bf16(false, a0.v, false, b1.v, (short)0, acc[0][1], false, false);
        acc[1][0] = __builtin_amdgcn_wmma_f32_16x16x32_bf16(false, a1.v, false, b0.v, (short)0, acc[1][0], false, false);
        acc[1][1] = __builtin_amdgcn_wmma_f32_16x16x32_bf16(false, a1.v, false, b1.v, (short)0, acc[1][1], false, false);
        __syncthreads();
    }
    int n = lane & 15, mb = (lane >> 4) << 3;
    #pragma unroll
    for (int mf = 0; mf < 2; mf++)
        #pragma unroll
        for (int nf = 0; nf < 2; nf++) {
            float* cp = C + (size_t)(mblk + mi * 32 + mf * 16 + mb) * N + nblk + ni * 32 + nf * 16 + n;
            #pragma unroll
            for (int v = 0; v < 8; v++) cp[(size_t)v * N] = acc[mf][nf][v];
        }
}

// ---------------------------------------------------------------- causal conv1d + silu + rope
__global__ void k_conv(const float* __restrict__ zx, const float* __restrict__ cw,
                       const float* __restrict__ cb, float* __restrict__ xbc) {
    int idx = blockIdx.x * 256 + threadIdx.x;
    if (idx >= BL * CONVDIM) return;
    int ch  = idx % CONVDIM;
    int row = idx / CONVDIM;
    int l   = row % SEQL;
    int b   = row / SEQL;

    auto conv1 = [&](int c) -> float {
        float s = cb[c];
        #pragma unroll
        for (int i = 0; i < 4; i++) {
            int ll = l - 3 + i;
            if (ll >= 0)
                s += zx[(size_t)(b * SEQL + ll) * DPROJP + DINNER + c] * cw[c * 4 + i];
        }
        return siluf(s);
    };

    float v = conv1(ch);
    int base = -1, m = 0;
    if (ch >= DINNER && ch < DINNER + DSTATE)       { base = DINNER;          m = ch - DINNER; }
    else if (ch >= DINNER + DSTATE)                 { base = DINNER + DSTATE; m = ch - DINNER - DSTATE; }
    if (base >= 0 && m < 64) {
        int j = (m < 32) ? m : m - 32;
        float freq = __powf(10000.f, -(float)j / 32.f);
        float ang  = (float)l * freq;
        float cv = __cosf(ang), sv = __sinf(ang);
        float vp = conv1(base + ((m < 32) ? m + 32 : m - 32));
        v = (m < 32) ? (v * cv - vp * sv) : (vp * sv + v * cv);
    }
    xbc[(size_t)row * CONVDIM + ch] = v;
}

// ---------------------------------------------------------------- dt = softplus(dt + bias)
__global__ void k_dt(const float* __restrict__ zx, const float* __restrict__ dt_bias,
                     float* __restrict__ dtp) {
    int idx = blockIdx.x * 256 + threadIdx.x;
    if (idx >= BL * NHEADSN) return;
    int h = idx % NHEADSN, row = idx / NHEADSN;
    float v = zx[(size_t)row * DPROJP + DINNER + CONVDIM + h] + dt_bias[h];
    dtp[idx] = (v > 20.f) ? v : log1pf(__expf(v));
}

// ---------------------------------------------------------------- intra-chunk SSD
// CB = C·B^T done with WMMA (waves 0-3); Yd and chunk states in VALU.
__global__ void k_ssd_intra(const float* __restrict__ xbc, const float* __restrict__ dtp,
                            const float* __restrict__ A_log, const float* __restrict__ Dparam,
                            float* __restrict__ Y, float* __restrict__ states,
                            float* __restrict__ dacs) {
    int idx = blockIdx.x;
    int h = idx % NHEADSN;
    int c = (idx / NHEADSN) % NCHUNK;
    int b = idx / (NHEADSN * NCHUNK);
    int row0 = b * SEQL + c * CHUNKL;

    __shared__ float Bs[CHUNKL][DSTP];
    __shared__ float Cs[CHUNKL][DSTP];
    __shared__ float xs[CHUNKL][HEADD + 4];
    __shared__ float att[CHUNKL][CHUNKL];
    __shared__ float dts[CHUNKL], cs[CHUNKL], coef[CHUNKL];

    int t = threadIdx.x;
    int w = t >> 5, lane = t & 31;
    for (int e = t; e < CHUNKL * DSTATE; e += 256) {
        int l = e >> 7, n = e & 127;
        Bs[l][n] = xbc[(size_t)(row0 + l) * CONVDIM + DINNER + n];
        Cs[l][n] = xbc[(size_t)(row0 + l) * CONVDIM + DINNER + DSTATE + n];
    }
    for (int e = t; e < CHUNKL * HEADD; e += 256) {
        int l = e >> 6, p = e & 63;
        xs[l][p] = xbc[(size_t)(row0 + l) * CONVDIM + h * HEADD + p];
    }
    if (t < CHUNKL) dts[t] = dtp[(size_t)(row0 + t) * NHEADSN + h];
    __syncthreads();
    if (t == 0) {
        float An = -__expf(A_log[h]);
        float acc = 0.f;
        for (int l = 0; l < CHUNKL; l++) { acc += dts[l] * An; cs[l] = acc; }
    }
    __syncthreads();
    if (t < CHUNKL) {
        coef[t] = dts[t] * __expf(cs[CHUNKL - 1] - cs[t]);
        dacs[(size_t)idx * CHUNKL + t] = cs[t];
    }
    __syncthreads();

    if (w < 4) {                                  // CB via WMMA: 4 tiles of 16x16
        int i0 = (w >> 1) << 4, j0 = (w & 1) << 4;
        int r = lane & 15, koff = (lane >> 4) << 3;
        f32x8 acc = {};
        #pragma unroll
        for (int kt = 0; kt < 4; kt++) {
            int kk = kt * 32 + koff;
            FragS fa, fb;
            #pragma unroll
            for (int q = 0; q < 8; q++) {
                fa.h[q]     = f32_to_bf16(Cs[i0 + r][kk + q]);
                fa.h[8 + q] = f32_to_bf16(Cs[i0 + r][kk + 16 + q]);
                fb.h[q]     = f32_to_bf16(Bs[j0 + r][kk + q]);
                fb.h[8 + q] = f32_to_bf16(Bs[j0 + r][kk + 16 + q]);
            }
            acc = __builtin_amdgcn_wmma_f32_16x16x32_bf16(false, fa.v, false, fb.v,
                                                          (short)0, acc, false, false);
        }
        int n = lane & 15, mb = (lane >> 4) << 3;
        #pragma unroll
        for (int v = 0; v < 8; v++) {
            int i = i0 + mb + v, j = j0 + n;
            att[i][j] = (j <= i) ? acc[v] * __expf(cs[i] - cs[j]) * dts[j] : 0.f;
        }
    }
    __syncthreads();

    float Dp = Dparam[h];
    for (int e = t; e < CHUNKL * HEADD; e += 256) {
        int i = e >> 6, p = e & 63;
        float s = Dp * xs[i][p];
        for (int j = 0; j <= i; j++) s += att[i][j] * xs[j][p];
        Y[(size_t)(row0 + i) * DINNER + h * HEADD + p] = s;
    }
    float* st = states + (size_t)idx * (HEADD * DSTATE);
    for (int e = t; e < HEADD * DSTATE; e += 256) {
        int p = e >> 7, n = e & 127;
        float s = 0.f;
        for (int l = 0; l < CHUNKL; l++) s += coef[l] * Bs[l][n] * xs[l][p];
        st[e] = s;
    }
}

// ---------------------------------------------------------------- inter-chunk scan (in place)
__global__ void k_ssd_scan(float* __restrict__ states, const float* __restrict__ dacs) {
    int bh = blockIdx.x;                 // b*NHEADS + h
    int h = bh % NHEADSN, b = bh / NHEADSN;
    int t = threadIdx.x;
    float run[32];
    #pragma unroll
    for (int k = 0; k < 32; k++) run[k] = 0.f;
    for (int c = 0; c < NCHUNK; c++) {
        size_t idx = ((size_t)(b * NCHUNK + c) * NHEADSN + h);
        float cd = __expf(dacs[idx * CHUNKL + CHUNKL - 1]);
        float* st = states + idx * (HEADD * DSTATE);
        #pragma unroll
        for (int k = 0; k < 32; k++) {
            int e = t + k * 256;
            float s = st[e];
            st[e] = run[k];
            run[k] = run[k] * cd + s;
        }
    }
}

// ---------------------------------------------------------------- Yo (WMMA) + gate -> bf16 yv
// prev state tile (64x128 f32, 32KB) pulled into LDS via Tensor Data Mover with
// LDS padding (4 dwords per 128) for bank de-conflict; Yo = C·prev^T on WMMA.
__global__ void k_ssd_inter(const float* __restrict__ xbc, const float* __restrict__ states,
                            const float* __restrict__ dacs, const float* __restrict__ Y,
                            const float* __restrict__ zx, unsigned short* __restrict__ yv) {
    int idx = blockIdx.x;
    int h = idx % NHEADSN;
    int c = (idx / NHEADSN) % NCHUNK;
    int b = idx / (NHEADSN * NCHUNK);
    int row0 = b * SEQL + c * CHUNKL;

    __shared__ float ps[HEADD][DSTP];     // prev state (TDM destination, padded rows)
    __shared__ float Cs[CHUNKL][DSTP];
    __shared__ float ex[CHUNKL];

    int t = threadIdx.x;
    int w = t >> 5, lane = t & 31;
    const float* st = states + (size_t)idx * (HEADD * DSTATE);

    if (w == 0) {
        // D# group0: count=1, lds_addr, global_addr, type=2
        unsigned long long ga = (unsigned long long)(const void*)st;
        unsigned galo = __builtin_amdgcn_readfirstlane((unsigned)ga);
        unsigned gahi = __builtin_amdgcn_readfirstlane((unsigned)(ga >> 32));
        unsigned lofs = __builtin_amdgcn_readfirstlane((unsigned)(unsigned long long)(const void*)&ps[0][0]);
        u32x4 g0 = { 1u, lofs, galo, (gahi & 0x01FFFFFFu) | 0x80000000u };
        // group1: data_size=4B, pad_enable, pad_interval=128 dwords, pad_amount=4 dwords,
        // tensor_dim0=8192, tensor_dim1=1, tile_dim0=8192, tile_dim1=1, stride0=8192
        u32x8 g1 = { 0x07920000u, 0x20000000u, 0x00010000u, 0x20000000u,
                     1u, 8192u, 0x20000000u, 0u };
        asm volatile("tensor_load_to_lds %0, %1" :: "s"(g0), "s"(g1) : "memory");
        asm volatile("s_wait_tensorcnt 0x0" ::: "memory");
    }
    for (int e = t; e < CHUNKL * DSTATE; e += 256)
        Cs[e >> 7][e & 127] = xbc[(size_t)(row0 + (e >> 7)) * CONVDIM + DINNER + DSTATE + (e & 127)];
    if (t < CHUNKL) ex[t] = __expf(dacs[(size_t)idx * CHUNKL + t]);
    __syncthreads();

    // Yo tiles: 8 waves -> (i0 in {0,16}) x (p0 in {0,16,32,48})
    int i0 = (w >> 2) << 4;
    int p0 = (w & 3) << 4;
    int r = lane & 15, koff = (lane >> 4) << 3;
    f32x8 acc = {};
    #pragma unroll
    for (int kt = 0; kt < 4; kt++) {
        int kk = kt * 32 + koff;
        FragS fa, fb;
        #pragma unroll
        for (int q = 0; q < 8; q++) {
            fa.h[q]     = f32_to_bf16(Cs[i0 + r][kk + q]);
            fa.h[8 + q] = f32_to_bf16(Cs[i0 + r][kk + 16 + q]);
            fb.h[q]     = f32_to_bf16(ps[p0 + r][kk + q]);
            fb.h[8 + q] = f32_to_bf16(ps[p0 + r][kk + 16 + q]);
        }
        acc = __builtin_amdgcn_wmma_f32_16x16x32_bf16(false, fa.v, false, fb.v,
                                                      (short)0, acc, false, false);
    }
    int n = lane & 15, mb = (lane >> 4) << 3;
    #pragma unroll
    for (int v = 0; v < 8; v++) {
        int i = i0 + mb + v, p = p0 + n;
        float val = Y[(size_t)(row0 + i) * DINNER + h * HEADD + p] + ex[i] * acc[v];
        float z = zx[(size_t)(row0 + i) * DPROJP + h * HEADD + p];
        yv[(size_t)(row0 + i) * DINNER + h * HEADD + p] = f32_to_bf16(val * siluf(z));
    }
}

// ---------------------------------------------------------------- final residual rmsnorm
__global__ void k_final(const float* __restrict__ x, const float* __restrict__ om,
                        const float* __restrict__ scale, float* __restrict__ out) {
    int row = blockIdx.x;
    const float* o = om + (size_t)row * DMODEL;
    __shared__ float red[8];
    float ss = 0.f;
    for (int d = threadIdx.x; d < DMODEL; d += 256) { float v = o[d]; ss += v * v; }
    for (int off = 16; off > 0; off >>= 1) ss += __shfl_down(ss, off, 32);
    int wid = threadIdx.x >> 5, lane = threadIdx.x & 31;
    if (lane == 0) red[wid] = ss;
    __syncthreads();
    if (threadIdx.x == 0) {
        float t = 0.f;
        for (int i = 0; i < 8; i++) t += red[i];
        red[0] = rsqrtf(t / (float)DMODEL + 1e-6f);
    }
    __syncthreads();
    float r = red[0];
    for (int d = threadIdx.x; d < DMODEL; d += 256)
        out[(size_t)row * DMODEL + d] = x[(size_t)row * DMODEL + d] + o[d] * r * (1.0f + scale[d]);
}

// ---------------------------------------------------------------- launch
extern "C" void kernel_launch(void* const* d_in, const int* in_sizes, int n_in,
                              void* d_out, int out_size, void* d_ws, size_t ws_size,
                              hipStream_t stream) {
    const float* x          = (const float*)d_in[0];
    const float* pre_scale  = (const float*)d_in[1];
    const float* post_scale = (const float*)d_in[2];
    const float* W_in       = (const float*)d_in[3];
    const float* conv_w     = (const float*)d_in[4];
    const float* conv_b     = (const float*)d_in[5];
    const float* dt_bias    = (const float*)d_in[6];
    const float* A_log      = (const float*)d_in[7];
    const float* D_param    = (const float*)d_in[8];
    const float* W_out      = (const float*)d_in[9];
    float* out = (float*)d_out;

    char* ws = (char*)d_ws;
    size_t off = 0;
    auto alloc = [&](size_t bytes) -> char* {
        char* p = ws + off;
        off += (bytes + 255) & ~(size_t)255;
        return p;
    };
    unsigned short* yb16   = (unsigned short*)alloc((size_t)BL * DMODEL * 2);
    unsigned short* winT   = (unsigned short*)alloc((size_t)DPROJP * DMODEL * 2);
    unsigned short* woutT  = (unsigned short*)alloc((size_t)DMODEL * DINNER * 2);
    float*          zx     = (float*)alloc((size_t)BL * DPROJP * 4);
    float*          xbc    = (float*)alloc((size_t)BL * CONVDIM * 4);
    float*          dtp    = (float*)alloc((size_t)BL * NHEADSN * 4);
    float*          dacs   = (float*)alloc((size_t)BATCHN * NCHUNK * NHEADSN * CHUNKL * 4);
    float*          Ybuf   = (float*)alloc((size_t)BL * DINNER * 4);
    float*          states = (float*)alloc((size_t)BATCHN * NCHUNK * NHEADSN * HEADD * DSTATE * 4);
    unsigned short* yv     = (unsigned short*)alloc((size_t)BL * DINNER * 2);
    float*          outmm  = (float*)alloc((size_t)BL * DMODEL * 4);

    k_rmsnorm_in<<<BL, 256, 0, stream>>>(x, pre_scale, yb16);
    k_prep_win<<<(DPROJP * DMODEL + 255) / 256, 256, 0, stream>>>(W_in, winT);
    k_prep_wout<<<(DMODEL * DINNER + 255) / 256, 256, 0, stream>>>(W_out, woutT);

    k_gemm_bf16<<<dim3(DPROJP / 128, BL / 64), 256, 0, stream>>>(yb16, winT, zx, BL, DPROJP, DMODEL);

    k_conv<<<(BL * CONVDIM + 255) / 256, 256, 0, stream>>>(zx, conv_w, conv_b, xbc);
    k_dt<<<(BL * NHEADSN + 255) / 256, 256, 0, stream>>>(zx, dt_bias, dtp);

    k_ssd_intra<<<BATCHN * NCHUNK * NHEADSN, 256, 0, stream>>>(xbc, dtp, A_log, D_param,
                                                               Ybuf, states, dacs);
    k_ssd_scan<<<BATCHN * NHEADSN, 256, 0, stream>>>(states, dacs);
    k_ssd_inter<<<BATCHN * NCHUNK * NHEADSN, 256, 0, stream>>>(xbc, states, dacs, Ybuf, zx, yv);

    k_gemm_bf16<<<dim3(DMODEL / 128, BL / 64), 256, 0, stream>>>(yv, woutT, outmm, BL, DMODEL, DINNER);

    k_final<<<BL, 256, 0, stream>>>(x, outmm, post_scale, out);
}
